// QLinear_73864847556832
// MI455X (gfx1250) — matrix operand order repaired and checked by reading, not compiled, not run on hardware
//
#include <hip/hip_runtime.h>

// ---------------------------------------------------------------------------
// NF4 dequant + y = x @ qw^T for MI455X (gfx1250, wave32, WMMA bf16 + TDM)
// x[8192,4096] f32, weight[4096,4096] f32, max_val[M*N/64] f32, lut[16] f32
// d_out = [ y (8192*4096 f32) | qweight (4096*4096 f32) ]
// Fast path (ws >= 100.7MB): pre-convert x/qw to bf16 in d_ws, GEMM stages
// tiles via tensor_load_to_lds (TDM, TENSORcnt). Fallback: in-loop f32->bf16.
// ---------------------------------------------------------------------------

typedef __attribute__((ext_vector_type(16))) __bf16    v16bf;
typedef __attribute__((ext_vector_type(2)))  __bf16    v2bf;
typedef __attribute__((ext_vector_type(8)))  float     v8f;
typedef __attribute__((ext_vector_type(4)))  float     v4f;
typedef __attribute__((ext_vector_type(4)))  unsigned  v4u;
typedef __attribute__((ext_vector_type(8)))  int       v8i;
typedef __attribute__((ext_vector_type(4)))  int       v4i;

#define T_DIM 8192
#define N_DIM 4096
#define M_DIM 4096

#define BT 128            // token tile
#define BM 128            // output-column (m) tile
#define BK 64             // K step (2 WMMA K-slices)
#define LDS_STRIDE 72     // bf16 elems per LDS row (64 + 8 pad) -> 144 B, 16B-aligned

union FragBF16 {
  v16bf v;
  uint4 q[2];
};

__device__ __forceinline__ unsigned pack2_bf16(float a, float b) {
  // pairs into one dword -> backend selects v_cvt_pk_bf16_f32 (RNE)
  v2bf p;
  p.x = (__bf16)a;
  p.y = (__bf16)b;
  return __builtin_bit_cast(unsigned, p);
}
__device__ __forceinline__ uint2 pack4_bf16(v4f f) {
  return make_uint2(pack2_bf16(f.x, f.y), pack2_bf16(f.z, f.w));
}

// ---------------------------------------------------------------------------
// Kernel 1: blockwise NF4 quantize+dequantize. Writes qweight f32 (output);
// optionally also writes packed bf16 copy for the TDM GEMM path.
// ---------------------------------------------------------------------------
__global__ __launch_bounds__(256) void nf4_dequant_kernel(
    const float* __restrict__ w, const float* __restrict__ maxval,
    const float* __restrict__ lut, float* __restrict__ qw,
    unsigned short* __restrict__ qwb16, int n4) {
  __shared__ float slut[16];
  if (threadIdx.x < 16) slut[threadIdx.x] = lut[threadIdx.x];
  __syncthreads();

  int idx4 = blockIdx.x * 256 + threadIdx.x;   // float4 index
  if (idx4 >= n4) return;
  float scale = maxval[(idx4 * 4) >> 6];       // 64-element quant blocks
  float inv = 1.0f / scale;
  v4f wv = __builtin_nontemporal_load(reinterpret_cast<const v4f*>(w) + idx4);
  float in[4] = {wv.x, wv.y, wv.z, wv.w};
  float out[4];
#pragma unroll
  for (int e = 0; e < 4; ++e) {
    float t = in[e] * inv;
    float best = slut[0];
    float bd = __builtin_fabsf(t - slut[0]);
#pragma unroll
    for (int j = 1; j < 16; ++j) {
      float d = __builtin_fabsf(t - slut[j]);
      if (d < bd) { bd = d; best = slut[j]; }   // strict < keeps first tie (argmin)
    }
    out[e] = best * scale;
  }
  v4f ov; ov.x = out[0]; ov.y = out[1]; ov.z = out[2]; ov.w = out[3];
  reinterpret_cast<v4f*>(qw)[idx4] = ov;
  if (qwb16) reinterpret_cast<uint2*>(qwb16)[idx4] = pack4_bf16(ov);
}

// ---------------------------------------------------------------------------
// Kernel 1b: x f32 -> bf16 (one streaming pass; x f32 never read again)
// ---------------------------------------------------------------------------
__global__ __launch_bounds__(256) void f32_to_bf16_kernel(
    const float* __restrict__ in, unsigned short* __restrict__ out, int n8) {
  int i = blockIdx.x * 256 + threadIdx.x;      // 8 elements per thread
  if (i >= n8) return;
  v4f a = __builtin_nontemporal_load(reinterpret_cast<const v4f*>(in) + 2 * i);
  v4f b = __builtin_nontemporal_load(reinterpret_cast<const v4f*>(in) + 2 * i + 1);
  uint4 o = make_uint4(pack2_bf16(a.x, a.y), pack2_bf16(a.z, a.w),
                       pack2_bf16(b.x, b.y), pack2_bf16(b.z, b.w));
  reinterpret_cast<uint4*>(out)[i] = o;
}

// ---------------------------------------------------------------------------
// TDM descriptor issue: one tensor_load_to_lds moves a 128x64 bf16 tile
// (row stride N_DIM elems) into LDS with HW padding 128B->144B rows.
// ---------------------------------------------------------------------------
__device__ __forceinline__ void tdm_load_tile(const unsigned short* gbase,
                                              unsigned lds_addr) {
  unsigned long long ga = (unsigned long long)(uintptr_t)gbase;
  v4u g0;
  g0[0] = 1u;                                   // count=1, user descriptor
  g0[1] = lds_addr;                             // LDS byte address
  g0[2] = (unsigned)ga;                         // global_addr[31:0]
  g0[3] = ((unsigned)(ga >> 32) & 0x01FFFFFFu)  // global_addr[56:32]
          | (2u << 30);                         // type=2 ("image")
  v8i g1;
  g1[0] = (1 << 16)      // data_size = 2 bytes
        | (1 << 20)      // pad_enable
        | (4 << 22)      // pad_interval: 32 DWORDs (=128B, one tile row)
        | (3 << 25);     // pad_amount: 4 DWORDs (=16B) -> 144B LDS rows
  g1[1] = (N_DIM & 0xFFFF) << 16;               // tensor_dim0[15:0]
  g1[2] = (N_DIM >> 16) | (BT << 16);           // tensor_dim0[31:16] | tensor_dim1[15:0]
  g1[3] = (BK << 16);                           // tensor_dim1[31:16]=0 | tile_dim0=64
  g1[4] = BT;                                   // tile_dim1=128 | tile_dim2=0
  g1[5] = N_DIM;                                // tensor_dim0_stride[31:0]
  g1[6] = 0;                                    // stride[47:32]=0 | dim1_stride[15:0]=0
  g1[7] = 0;
  v4i g2 = {0, 0, 0, 0};
  v4i g3 = {0, 0, 0, 0};
  v8i g4 = {0, 0, 0, 0, 0, 0, 0, 0};            // 6-arg toolchain: extra group
  __builtin_amdgcn_tensor_load_to_lds(g0, g1, g2, g3, g4, 0);
}

// ---------------------------------------------------------------------------
// Shared WMMA compute: 2 K-slices x 8 v_wmma_f32_16x16x32_bf16 on one buffer
// ---------------------------------------------------------------------------
__device__ __forceinline__ void wmma_tile(const unsigned short* cx,
                                          const unsigned short* cw,
                                          int waveT, int waveM, int lrow, int lhi,
                                          v8f acc[2][4]) {
#pragma unroll
  for (int kk = 0; kk < BK; kk += 32) {
    FragBF16 a[2], b[4];
#pragma unroll
    for (int tt = 0; tt < 2; ++tt) {
      int row = waveT * 32 + tt * 16 + lrow;
      const char* base = (const char*)cx + row * (LDS_STRIDE * 2) + kk * 2 + lhi * 16;
      a[tt].q[0] = *reinterpret_cast<const uint4*>(base);
      a[tt].q[1] = *reinterpret_cast<const uint4*>(base + 32);
    }
#pragma unroll
    for (int mm = 0; mm < 4; ++mm) {
      int row = waveM * 64 + mm * 16 + lrow;
      const char* base = (const char*)cw + row * (LDS_STRIDE * 2) + kk * 2 + lhi * 16;
      b[mm].q[0] = *reinterpret_cast<const uint4*>(base);
      b[mm].q[1] = *reinterpret_cast<const uint4*>(base + 32);
    }
#pragma unroll
    for (int mm = 0; mm < 4; ++mm)
#pragma unroll
      for (int tt = 0; tt < 2; ++tt)
        acc[tt][mm] = __builtin_amdgcn_wmma_f32_16x16x32_bf16(
            false, a[tt].v, false, b[mm].v, (short)0, acc[tt][mm], false, false);
  }
}

__device__ __forceinline__ void store_epilogue(float* __restrict__ y, int t0, int m0,
                                               int waveT, int waveM, int lrow, int lhi,
                                               v8f acc[2][4]) {
#pragma unroll
  for (int tt = 0; tt < 2; ++tt)
#pragma unroll
    for (int mm = 0; mm < 4; ++mm) {
      int m = m0 + waveM * 64 + mm * 16 + lrow;
#pragma unroll
      for (int r = 0; r < 8; ++r) {
        int t = t0 + waveT * 32 + tt * 16 + r + lhi * 8;
        __builtin_nontemporal_store(acc[tt][mm][r], &y[(size_t)t * M_DIM + m]);
      }
    }
}

// ---------------------------------------------------------------------------
// Kernel 2a (fast path): bf16 operands in global; tiles staged by the Tensor
// Data Mover (wave 0 issues, TENSORcnt-synced), double-buffered LDS.
// ---------------------------------------------------------------------------
__global__ __launch_bounds__(256) void qgemm_wmma_tdm_kernel(
    const unsigned short* __restrict__ xb, const unsigned short* __restrict__ wb,
    float* __restrict__ y) {
  __shared__ unsigned short sX[2][BT * LDS_STRIDE];
  __shared__ unsigned short sW[2][BM * LDS_STRIDE];

  const int tid   = threadIdx.x;
  const int lane  = tid & 31;
  const int wid   = tid >> 5;
  const int lrow  = lane & 15;
  const int lhi   = lane >> 4;
  const int waveT = wid >> 1;
  const int waveM = wid & 1;

  const int t0 = blockIdx.y * BT;
  const int m0 = blockIdx.x * BM;

  const unsigned short* xtile = xb + (size_t)t0 * N_DIM;
  const unsigned short* wtile = wb + (size_t)m0 * N_DIM;

  v8f acc[2][4] = {};

  // prologue: DMA tile 0 into buffer 0
  if (wid == 0) {
    tdm_load_tile(xtile, (unsigned)(uintptr_t)&sX[0][0]);
    tdm_load_tile(wtile, (unsigned)(uintptr_t)&sW[0][0]);
    __builtin_amdgcn_s_wait_tensorcnt(0);
  }
  __syncthreads();

  int buf = 0;
  for (int k0 = 0; k0 < N_DIM; k0 += BK) {
    const bool more = (k0 + BK) < N_DIM;
    if (more && wid == 0) {   // DMA next tile into the other buffer
      tdm_load_tile(xtile + k0 + BK, (unsigned)(uintptr_t)&sX[buf ^ 1][0]);
      tdm_load_tile(wtile + k0 + BK, (unsigned)(uintptr_t)&sW[buf ^ 1][0]);
    }

    wmma_tile(sX[buf], sW[buf], waveT, waveM, lrow, lhi, acc);

    if (more && wid == 0) __builtin_amdgcn_s_wait_tensorcnt(0);
    __syncthreads();
    buf ^= 1;
  }

  store_epilogue(y, t0, m0, waveT, waveM, lrow, lhi, acc);
}

// ---------------------------------------------------------------------------
// Kernel 2b (fallback): f32 operands, in-loop bf16 conversion
// ---------------------------------------------------------------------------
__global__ __launch_bounds__(256) void qgemm_wmma_kernel(
    const float* __restrict__ x, const float* __restrict__ qw,
    float* __restrict__ y) {
  __shared__ unsigned short sX[2][BT * LDS_STRIDE];
  __shared__ unsigned short sW[2][BM * LDS_STRIDE];

  const int tid   = threadIdx.x;
  const int lane  = tid & 31;
  const int wid   = tid >> 5;
  const int lrow  = lane & 15;
  const int lhi   = lane >> 4;
  const int waveT = wid >> 1;
  const int waveM = wid & 1;

  const int t0 = blockIdx.y * BT;
  const int m0 = blockIdx.x * BM;

  const int r0 = tid >> 4;
  const int c4 = tid & 15;
  const float* xbase = x  + (size_t)(t0 + r0) * N_DIM + c4 * 4;
  const float* wbase = qw + (size_t)(m0 + r0) * N_DIM + c4 * 4;
  const unsigned ldsoff = r0 * LDS_STRIDE + c4 * 4;

  v8f acc[2][4] = {};
  v4f xr[8], wr[8];

#pragma unroll
  for (int i = 0; i < 8; ++i) {
    xr[i] = *reinterpret_cast<const v4f*>(xbase + (size_t)i * 16 * N_DIM);
    wr[i] = *reinterpret_cast<const v4f*>(wbase + (size_t)i * 16 * N_DIM);
  }
#pragma unroll
  for (int i = 0; i < 8; ++i) {
    *reinterpret_cast<uint2*>(&sX[0][ldsoff + i * 16 * LDS_STRIDE]) = pack4_bf16(xr[i]);
    *reinterpret_cast<uint2*>(&sW[0][ldsoff + i * 16 * LDS_STRIDE]) = pack4_bf16(wr[i]);
  }
  __syncthreads();

  int buf = 0;
  for (int k0 = 0; k0 < N_DIM; k0 += BK) {
    const bool more = (k0 + BK) < N_DIM;
    if (more) {
#pragma unroll
      for (int i = 0; i < 8; ++i) {
        xr[i] = *reinterpret_cast<const v4f*>(xbase + k0 + BK + (size_t)i * 16 * N_DIM);
        wr[i] = *reinterpret_cast<const v4f*>(wbase + k0 + BK + (size_t)i * 16 * N_DIM);
      }
    }

    wmma_tile(sX[buf], sW[buf], waveT, waveM, lrow, lhi, acc);

    if (more) {
      unsigned short* px = sX[buf ^ 1];
      unsigned short* pw = sW[buf ^ 1];
#pragma unroll
      for (int i = 0; i < 8; ++i) {
        *reinterpret_cast<uint2*>(&px[ldsoff + i * 16 * LDS_STRIDE]) = pack4_bf16(xr[i]);
        *reinterpret_cast<uint2*>(&pw[ldsoff + i * 16 * LDS_STRIDE]) = pack4_bf16(wr[i]);
      }
    }
    __syncthreads();
    buf ^= 1;
  }

  store_epilogue(y, t0, m0, waveT, waveM, lrow, lhi, acc);
}

// ---------------------------------------------------------------------------
extern "C" void kernel_launch(void* const* d_in, const int* in_sizes, int n_in,
                              void* d_out, int out_size, void* d_ws, size_t ws_size,
                              hipStream_t stream) {
  (void)in_sizes; (void)n_in; (void)out_size;

  const float* x   = (const float*)d_in[0];
  const float* w   = (const float*)d_in[1];
  const float* mv  = (const float*)d_in[2];
  const float* lut = (const float*)d_in[3];

  float* y  = (float*)d_out;
  float* qw = y + (size_t)T_DIM * M_DIM;   // second tuple element, flat-concat

  const int n4 = (M_DIM * N_DIM) / 4;
  const size_t xb_elems = (size_t)T_DIM * N_DIM;
  const size_t wb_elems = (size_t)M_DIM * N_DIM;
  const size_t ws_need  = (xb_elems + wb_elems) * sizeof(unsigned short);

  if (ws_size >= ws_need) {
    // fast path: bf16 operands in workspace + TDM-staged GEMM
    unsigned short* xb = (unsigned short*)d_ws;
    unsigned short* wb = xb + xb_elems;
    nf4_dequant_kernel<<<dim3((n4 + 255) / 256), dim3(256), 0, stream>>>(
        w, mv, lut, qw, wb, n4);
    int n8 = (int)(xb_elems / 8);
    f32_to_bf16_kernel<<<dim3((n8 + 255) / 256), dim3(256), 0, stream>>>(x, xb, n8);
    qgemm_wmma_tdm_kernel<<<dim3(M_DIM / BM, T_DIM / BT), dim3(256), 0, stream>>>(
        xb, wb, y);
  } else {
    // fallback: f32 operands, in-loop conversion
    nf4_dequant_kernel<<<dim3((n4 + 255) / 256), dim3(256), 0, stream>>>(
        w, mv, lut, qw, nullptr, n4);
    qgemm_wmma_kernel<<<dim3(M_DIM / BM, T_DIM / BT), dim3(256), 0, stream>>>(
        x, qw, y);
  }
}
